// RINCE_75419625718616
// MI455X (gfx1250) — compile-verified, matrix operand order checked by reading, not run on hardware
//
#include <hip/hip_runtime.h>
#include <hip/hip_bf16.h>

typedef __attribute__((ext_vector_type(16))) _Float16 v16h;
typedef __attribute__((ext_vector_type(8)))  float    v8f;

#define TILE_M  16
#define CHUNK_K 32
#define SEG     4

// exp(x/T1) = exp(2x) = 2^(x * 2*log2(e))
#define TWO_LOG2E 2.8853900817779268f

__device__ __forceinline__ float fast_exp2(float x) { return __builtin_amdgcn_exp2f(x); }
__device__ __forceinline__ float fast_ln(float x)   { return __builtin_amdgcn_logf(x) * 0.69314718055994531f; }

// One wave per block. Each wave owns a 16-row tile and one column segment.
// WMMA computes, per 16x32 fp16 chunk:  D[m][0] += sum_k e1,  D[m][1] += mask * sum_k e1
// and (only when the uniform mask bit is set)  D2[m][1] += sum_k e1^2  (== e2 sum).
__global__ __launch_bounds__(32)
void rince_tile_kernel(const float* __restrict__ sim,
                       const int*   __restrict__ classes,
                       float* __restrict__ wsAll,
                       float* __restrict__ wsSame,
                       float* __restrict__ wsS2,
                       int N, int PPG, int rowTiles, int segW)
{
    const int tile     = blockIdx.x % rowTiles;
    const int seg      = blockIdx.x / rowTiles;
    const int rowBase  = tile * TILE_M;
    const int colStart = seg * segW;

    const int lane = threadIdx.x;
    const int n    = lane & 15;   // A-matrix row M held by this lane; C/D column N
    const int half = lane >> 4;

    // Row class is uniform over the 16-row tile (tiles never cross a PPG=64 group).
    const int rcls = classes[rowBase / PPG];

    // Uniform bitmap: which column groups in this segment share the row class.
    const int gps = segW / PPG;   // 32 groups per segment here
    unsigned int bitmap = 0u;
    for (int g = 0; g < gps; ++g)
        bitmap |= (classes[colStart / PPG + g] == rcls) ? (1u << g) : 0u;

    // Constant-per-lane B matrices (B[k][n] depends only on n = lane&15):
    //   b1_no : col0 = 1 (total e1)             ; col1 = 0
    //   b1_yes: col0 = 1, col1 = 1 (same-class chunk)
    //   b2    : col1 = 1 only (same-class e2)
    const _Float16 one = (_Float16)1.0f, zero = (_Float16)0.0f;
    v16h b1_no, b1_yes, b2;
#pragma unroll
    for (int i = 0; i < 16; ++i) {
        b1_no[i]  = (n == 0) ? one : zero;
        b1_yes[i] = (n <= 1) ? one : zero;
        b2[i]     = (n == 1) ? one : zero;
    }

    v8f acc1 = {};   // D[m][0] = sum e1 (all), D[m][1] = sum e1 (same class)
    v8f acc2 = {};   // D[m][1] = sum e2 (same class)

    const float* rowp = sim + (size_t)(rowBase + n) * N;

    for (int c = colStart; c < colStart + segW; c += CHUNK_K) {
        const int mb = (bitmap >> ((c - colStart) / PPG)) & 1;   // wave-uniform

        // lanes L and L+16 jointly cover the 32 columns of row (rowBase + L)
        const float4* p = (const float4*)(rowp + c + half * 16);
        float4 f0 = p[0], f1 = p[1], f2 = p[2], f3 = p[3];
        float t[16] = { f0.x, f0.y, f0.z, f0.w,
                        f1.x, f1.y, f1.z, f1.w,
                        f2.x, f2.y, f2.z, f2.w,
                        f3.x, f3.y, f3.z, f3.w };

        v16h a1;
#pragma unroll
        for (int i = 0; i < 16; ++i) {
            t[i]  = fast_exp2(t[i] * TWO_LOG2E);   // e1 = exp(sim/T1)
            a1[i] = (_Float16)t[i];
        }
        acc1 = __builtin_amdgcn_wmma_f32_16x16x32_f16(
                   false, a1, false, mb ? b1_yes : b1_no,
                   (short)0, acc1, false, false);

        if (mb) {                                   // uniform branch: EXEC stays all-1s
            v16h a2;
#pragma unroll
            for (int i = 0; i < 16; ++i)
                a2[i] = (_Float16)(t[i] * t[i]);    // e2 = e1^2 = exp(sim/T2)
            acc2 = __builtin_amdgcn_wmma_f32_16x16x32_f16(
                       false, a2, false, b2, (short)0, acc2, false, false);
        }
    }

    // C/D layout: lane holds column n = lane&15; VGPR r holds row m = r + 8*half.
    const size_t base = (size_t)seg * N + rowBase + 8 * half;
    if (n == 0) {
#pragma unroll
        for (int r = 0; r < 8; ++r) wsAll[base + r] = acc1[r];
    } else if (n == 1) {
#pragma unroll
        for (int r = 0; r < 8; ++r) { wsSame[base + r] = acc1[r]; wsS2[base + r] = acc2[r]; }
    }
}

__global__ __launch_bounds__(256)
void rince_row_loss(const float* __restrict__ sim,
                    const int*   __restrict__ posIdx,
                    const float* __restrict__ wsAll,
                    const float* __restrict__ wsSame,
                    const float* __restrict__ wsS2,
                    float* __restrict__ blockPart,
                    int N)
{
    const int row = blockIdx.x * blockDim.x + threadIdx.x;
    float loss = 0.0f;
    if (row < N) {
        float sa = 0.f, ss = 0.f, s2 = 0.f;
#pragma unroll
        for (int s = 0; s < SEG; ++s) {
            sa += wsAll [(size_t)s * N + row];
            ss += wsSame[(size_t)s * N + row];
            s2 += wsS2  [(size_t)s * N + row];
        }
        const float x   = sim[(size_t)row * N + posIdx[row]];
        const float pe  = fast_exp2(x * TWO_LOG2E);           // single positive e1
        const float neg = sa - pe;                            // first_neg_denom
        const float l1  = fast_ln(ss + neg) - 2.0f * x;       // -log(pe/(pos_denom+neg)); ln(pe)=2x
        const float l2  = fast_ln(s2 + neg) - fast_ln(s2);    // -log(s2/(s2+neg))
        loss = l1 + l2;
    }
    __shared__ float red[256];
    red[threadIdx.x] = loss;
    __syncthreads();
    for (int s = 128; s > 0; s >>= 1) {
        if (threadIdx.x < s) red[threadIdx.x] += red[threadIdx.x + s];
        __syncthreads();
    }
    if (threadIdx.x == 0) blockPart[blockIdx.x] = red[0];
}

__global__ void rince_final(const float* __restrict__ blockPart,
                            float* __restrict__ out, int nPart, int N)
{
    if (threadIdx.x == 0) {
        float s = 0.f;
        for (int i = 0; i < nPart; ++i) s += blockPart[i];
        out[0] = s / (float)N;
    }
}

extern "C" void kernel_launch(void* const* d_in, const int* in_sizes, int n_in,
                              void* d_out, int out_size, void* d_ws, size_t ws_size,
                              hipStream_t stream)
{
    const float* sim     = (const float*)d_in[0];   // [1, N, N] fp32
    const int*   posIdx  = (const int*)  d_in[1];   // [N]
    const int*   classes = (const int*)  d_in[2];   // [G]
    // d_in[3] = n_points: constant N/G per group; grouping derived from sizes.

    const int N   = in_sizes[1];      // 8192
    const int G   = in_sizes[2];      // 128
    const int PPG = N / G;            // 64

    float* wsAll     = (float*)d_ws;
    float* wsSame    = wsAll  + (size_t)SEG * N;
    float* wsS2      = wsSame + (size_t)SEG * N;
    float* blockPart = wsS2   + (size_t)SEG * N;

    const int rowTiles = N / TILE_M;  // 512
    const int segW     = N / SEG;     // 2048

    rince_tile_kernel<<<dim3(rowTiles * SEG), 32, 0, stream>>>(
        sim, classes, wsAll, wsSame, wsS2, N, PPG, rowTiles, segW);

    const int nBlocks2 = (N + 255) / 256;   // 32
    rince_row_loss<<<dim3(nBlocks2), 256, 0, stream>>>(
        sim, posIdx, wsAll, wsSame, wsS2, blockPart, N);

    rince_final<<<dim3(1), 32, 0, stream>>>(blockPart, (float*)d_out, nBlocks2, N);
}
